// ROIHead_43946105372768
// MI455X (gfx1250) — matrix operand order repaired and checked by reading, not compiled, not run on hardware
//
#include <hip/hip_runtime.h>
#include <math.h>

// ---------------- Types for CDNA5 WMMA ----------------
typedef __attribute__((ext_vector_type(16))) __bf16 v16bf;
typedef __attribute__((ext_vector_type(8)))  float  v8f;
typedef __attribute__((ext_vector_type(4)))  int    v4i;

// ---------------- Optional CDNA5 async-to-LDS path ----------------
#if defined(__has_builtin)
#if __has_builtin(__builtin_amdgcn_global_load_async_to_lds_b128) && \
    __has_builtin(__builtin_amdgcn_s_wait_asynccnt)
#define HAVE_ASYNC_LDS 1
#endif
#endif

#ifdef HAVE_ASYNC_LDS
typedef __attribute__((address_space(1))) v4i as1_v4i;
typedef __attribute__((address_space(3))) v4i as3_v4i;
__device__ __forceinline__ void async_copy16(const void* g, void* l) {
  // 16B global -> LDS, tracked by ASYNCcnt
  __builtin_amdgcn_global_load_async_to_lds_b128((as1_v4i*)g, (as3_v4i*)l, 0, 0);
}
#endif

// ---------------- Problem constants ----------------
#define NPROP   1000
#define NPAD    1024          // 64 tiles of 16 rows (full 8-wave blocks)
#define FIN     12544         // 256*7*7, = 392 * 32
#define H1      1024
#define NCLS    21
#define NFG     20
#define NCAND   20000         // NPROP * NFG
#define MAXDET  100
#define NEG_INF (-__builtin_inff())

// ---------------- Helpers ----------------
__device__ __forceinline__ unsigned short f32_to_bf16(float f) {
  unsigned int u = __float_as_uint(f);
  u += 0x7FFFu + ((u >> 16) & 1u);    // round-to-nearest-even
  return (unsigned short)(u >> 16);
}

// A-matrix fragment layout (16-bit, 16x32 tile, wave32):
//   lanes 0-15: M=0..15, K in {0..7, 16..23}; lanes 16-31: K in {8..15, 24..31}
__device__ __forceinline__ size_t packA_index(int m, int k, int KT) {
  int mt = m >> 4, kt = k >> 5, kb = k & 31;
  int lane = (m & 15) | (((kb >> 3) & 1) << 4);
  int e = (kb & 7) | (((kb >> 4) & 1) << 3);
  return (((size_t)(mt * KT + kt) * 32) + (size_t)lane) * 16 + (size_t)e;
}

// B-matrix fragment layout (16-bit, 32x16 tile, wave32):
//   lane = column (n&15) + 16*((K>>3)&1); elems: K in {0..7,16..23} / {8..15,24..31}
__device__ __forceinline__ size_t packB_index(int k, int n, int NT) {
  int kt = k >> 5, nt = n >> 4;
  int lane = (n & 15) | (((k >> 3) & 1) << 4);
  int e = (k & 7) | (((k >> 4) & 1) << 3);
  return (((size_t)(kt * NT + nt) * 32) + (size_t)lane) * 16 + (size_t)e;
}

// ---------------- 1. ROI-align, writes bf16 A-fragments directly ----------------
__global__ void roi_align_pack_kernel(const float* __restrict__ fm,
                                      const float* __restrict__ proposals,
                                      unsigned short* __restrict__ Apk) {
  int t = blockIdx.x * blockDim.x + threadIdx.x;
  if (t >= NPAD * FIN) return;
  int n   = t / FIN;
  int col = t - n * FIN;

  float val = 0.0f;
  if (n < NPROP) {
    int c  = col / 49;
    int rr = col - c * 49;
    int ph = rr / 7;
    int pw = rr - ph * 7;

    float bx0 = proposals[n * 4 + 0] * 0.125f - 0.5f;
    float by0 = proposals[n * 4 + 1] * 0.125f - 0.5f;
    float bx1 = proposals[n * 4 + 2] * 0.125f - 0.5f;
    float by1 = proposals[n * 4 + 3] * 0.125f - 0.5f;
    float bw = (bx1 - bx0) * (1.0f / 7.0f);
    float bh = (by1 - by0) * (1.0f / 7.0f);

    const float* f = fm + (size_t)c * 10000;
    float acc = 0.0f;
#pragma unroll
    for (int sy = 0; sy < 2; ++sy) {
#pragma unroll
      for (int sx = 0; sx < 2; ++sx) {
        float X = bx0 + ((float)pw + ((float)sx + 0.5f) * 0.5f) * bw;
        float Y = by0 + ((float)ph + ((float)sy + 0.5f) * 0.5f) * bh;
        bool vx = (X >= -1.0f) && (X <= 100.0f);
        bool vy = (Y >= -1.0f) && (Y <= 100.0f);
        float xc = fminf(fmaxf(X, 0.0f), 99.0f);
        float yc = fminf(fmaxf(Y, 0.0f), 99.0f);
        float x0f = floorf(xc), y0f = floorf(yc);
        int ix0 = (int)x0f, iy0 = (int)y0f;
        int ix1 = min(ix0 + 1, 99), iy1 = min(iy0 + 1, 99);
        float wx1 = xc - x0f, wx0 = 1.0f - wx1;
        float wy1 = yc - y0f, wy0 = 1.0f - wy1;
        float v = wy0 * (wx0 * f[iy0 * 100 + ix0] + wx1 * f[iy0 * 100 + ix1]) +
                  wy1 * (wx0 * f[iy1 * 100 + ix0] + wx1 * f[iy1 * 100 + ix1]);
        if (!(vx && vy)) v = 0.0f;
        acc += v;
      }
    }
    val = acc * 0.25f;
  }
  Apk[packA_index(n, col, FIN / 32)] = f32_to_bf16(val);
}

// ---------------- 2. Weight fp32 -> bf16 B-fragment pack ----------------
__global__ void pack_b_kernel(const float* __restrict__ src,
                              unsigned short* __restrict__ dst,
                              int K, int N) {
  int idx = blockIdx.x * blockDim.x + threadIdx.x;
  if (idx >= K * N) return;
  int k = idx / N;
  int n = idx - k * N;
  dst[packB_index(k, n, N >> 4)] = f32_to_bf16(src[idx]);
}

// ---------------- 3. bf16 WMMA GEMM, LDS-staged B, fused bias + ReLU ----------------
// Block = 8 waves computing 128(M) x 64(N). Per K-step the block stages the shared
// 4KB B slab (4 N-tiles) into LDS (async-to-LDS when available, double buffered);
// each wave does 4 x v_wmma_f32_16x16x32_bf16 against its private A fragment.
__global__ __launch_bounds__(256)
void gemm_wmma_kernel(const unsigned short* __restrict__ Apk,
                      const unsigned short* __restrict__ Bpk,
                      const float* __restrict__ bias,
                      int Kt, int Nt,
                      unsigned short* __restrict__ outPk, int outKT,
                      float* __restrict__ outF32, int ldo,
                      int do_relu) {
  __shared__ __align__(32) unsigned short Bs[2][2048];   // 2 x 4KB

  int tid    = threadIdx.x;
  int lane   = tid & 31;
  int waveId = tid >> 5;
  int nGroups = Nt >> 2;                       // 4 N-tiles per block
  int mGroup  = blockIdx.x / nGroups;
  int nGroup  = blockIdx.x - mGroup * nGroups;
  int mt      = mGroup * 8 + waveId;           // NPAD/16 is a multiple of 8
  int ntBase  = nGroup * 4;

  v8f acc0 = {}, acc1 = {}, acc2 = {}, acc3 = {};

  const unsigned short* aBase = Apk + (size_t)mt * Kt * 512 + (size_t)lane * 16;
  const unsigned short* bBase = Bpk + (size_t)ntBase * 512;   // + kt*Nt*512

  // ---- prologue: stage kt = 0 into buffer 0 ----
#ifdef HAVE_ASYNC_LDS
  async_copy16(bBase + (size_t)tid * 8, &Bs[0][tid * 8]);
  __builtin_amdgcn_s_wait_asynccnt(0);
  __syncthreads();
#else
  {
    uint4 r = *(const uint4*)(bBase + (size_t)tid * 8);
    *(uint4*)(&Bs[0][tid * 8]) = r;
  }
  __syncthreads();
#endif

  for (int kt = 0; kt < Kt; ++kt) {
    int cur = kt & 1;
    int nxt = cur ^ 1;
    bool more = (kt + 1 < Kt);

#ifdef HAVE_ASYNC_LDS
    if (more)
      async_copy16(bBase + (size_t)(kt + 1) * Nt * 512 + (size_t)tid * 8,
                   &Bs[nxt][tid * 8]);
#else
    uint4 rnext = {};
    if (more)
      rnext = *(const uint4*)(bBase + (size_t)(kt + 1) * Nt * 512 + (size_t)tid * 8);
#endif

    const unsigned short* ap = aBase + (size_t)kt * 512;
    __builtin_prefetch(ap + 512, 0, 3);        // next A tile into near cache
    v16bf a  = *(const v16bf*)ap;
    v16bf b0 = *(const v16bf*)(&Bs[cur][0 * 512 + lane * 16]);
    v16bf b1 = *(const v16bf*)(&Bs[cur][1 * 512 + lane * 16]);
    v16bf b2 = *(const v16bf*)(&Bs[cur][2 * 512 + lane * 16]);
    v16bf b3 = *(const v16bf*)(&Bs[cur][3 * 512 + lane * 16]);

    acc0 = __builtin_amdgcn_wmma_f32_16x16x32_bf16(false, a, false, b0,
                                                   (short)0, acc0, false, false);
    acc1 = __builtin_amdgcn_wmma_f32_16x16x32_bf16(false, a, false, b1,
                                                   (short)0, acc1, false, false);
    acc2 = __builtin_amdgcn_wmma_f32_16x16x32_bf16(false, a, false, b2,
                                                   (short)0, acc2, false, false);
    acc3 = __builtin_amdgcn_wmma_f32_16x16x32_bf16(false, a, false, b3,
                                                   (short)0, acc3, false, false);

#ifdef HAVE_ASYNC_LDS
    __builtin_amdgcn_s_wait_asynccnt(0);
    __syncthreads();
#else
    __syncthreads();
    if (more) *(uint4*)(&Bs[nxt][tid * 8]) = rnext;
    __syncthreads();
#endif
  }

  // Epilogue: D-fragment layout (f32 16x16): VGPR r, lanes0-15 -> M=r, lanes16-31 -> M=8+r
  int mrow  = mt * 16 + ((lane >> 4) << 3);
  int ncol0 = ntBase * 16 + (lane & 15);
  float bs0 = bias[ncol0];
  float bs1 = bias[ncol0 + 16];
  float bs2 = bias[ncol0 + 32];
  float bs3 = bias[ncol0 + 48];
#pragma unroll
  for (int r = 0; r < 8; ++r) {
    int m = mrow + r;
    float v0 = acc0[r] + bs0;
    float v1 = acc1[r] + bs1;
    float v2 = acc2[r] + bs2;
    float v3 = acc3[r] + bs3;
    if (do_relu) {
      v0 = fmaxf(v0, 0.0f); v1 = fmaxf(v1, 0.0f);
      v2 = fmaxf(v2, 0.0f); v3 = fmaxf(v3, 0.0f);
    }
    if (outF32) {
      outF32[(size_t)m * ldo + ncol0]      = v0;
      outF32[(size_t)m * ldo + ncol0 + 16] = v1;
      outF32[(size_t)m * ldo + ncol0 + 32] = v2;
      outF32[(size_t)m * ldo + ncol0 + 48] = v3;
    }
    if (outPk) {
      outPk[packA_index(m, ncol0,      outKT)] = f32_to_bf16(v0);
      outPk[packA_index(m, ncol0 + 16, outKT)] = f32_to_bf16(v1);
      outPk[packA_index(m, ncol0 + 32, outKT)] = f32_to_bf16(v2);
      outPk[packA_index(m, ncol0 + 48, outKT)] = f32_to_bf16(v3);
    }
  }
}

// ---------------- 4. Class / box heads (x2 row staged in LDS) ----------------
__global__ void head_kernel(const float* __restrict__ x2,
                            const float* __restrict__ wc, const float* __restrict__ bc,
                            const float* __restrict__ wb, const float* __restrict__ bbias,
                            float* __restrict__ cls, float* __restrict__ box) {
  __shared__ __align__(16) float xs[H1];
  int n = blockIdx.x;
  int tid = threadIdx.x;
#ifdef HAVE_ASYNC_LDS
  // 4KB row = 256 x 16B chunks; 128 threads x 2 async b128 copies
  async_copy16(x2 + (size_t)n * H1 + tid * 4,         &xs[tid * 4]);
  async_copy16(x2 + (size_t)n * H1 + (tid + 128) * 4, &xs[(tid + 128) * 4]);
  __builtin_amdgcn_s_wait_asynccnt(0);
  __syncthreads();
#else
  for (int i = tid; i < H1; i += blockDim.x) xs[i] = x2[(size_t)n * H1 + i];
  __syncthreads();
#endif
  int j = tid;
  if (j < NCLS) {
    float s = bc[j];
    for (int k = 0; k < H1; ++k) s += xs[k] * wc[k * NCLS + j];
    cls[n * NCLS + j] = s;
  } else if (j < NCLS + NCLS * 4) {
    int j2 = j - NCLS;
    float s = bbias[j2];
    for (int k = 0; k < H1; ++k) s += xs[k] * wb[k * (NCLS * 4) + j2];
    box[n * (NCLS * 4) + j2] = s;
  }
}

// ---------------- 5. Softmax + box decode + candidate init ----------------
__global__ void postproc_kernel(const float* __restrict__ cls,
                                const float* __restrict__ box,
                                const float* __restrict__ proposals,
                                float* __restrict__ scores_f,
                                float* __restrict__ boxes_f,
                                float* __restrict__ boxes_off,
                                float* __restrict__ cand) {
  int n = blockIdx.x * blockDim.x + threadIdx.x;
  if (n >= NPROP) return;
  float l[NCLS];
  float mx = -1e30f;
#pragma unroll
  for (int k = 0; k < NCLS; ++k) { l[k] = cls[n * NCLS + k]; mx = fmaxf(mx, l[k]); }
  float sum = 0.0f;
#pragma unroll
  for (int k = 0; k < NCLS; ++k) { l[k] = expf(l[k] - mx); sum += l[k]; }
  float inv = 1.0f / sum;

  float px0 = proposals[n * 4 + 0], py0 = proposals[n * 4 + 1];
  float px1 = proposals[n * 4 + 2], py1 = proposals[n * 4 + 3];
  float w = px1 - px0, h = py1 - py0;
  float cx = px0 + 0.5f * w, cy = py0 + 0.5f * h;

  for (int k = 1; k < NCLS; ++k) {
    int j = n * NFG + (k - 1);
    float s = l[k] * inv;
    const float* d = box + n * (NCLS * 4) + k * 4;
    float dx = d[0], dy = d[1], dw = d[2], dh = d[3];
    float pcx = dx * w + cx, pcy = dy * h + cy;
    float pw_ = expf(dw) * w, ph_ = expf(dh) * h;
    float b0 = pcx - 0.5f * pw_, b1 = pcy - 0.5f * ph_;
    float b2 = pcx + 0.5f * pw_, b3 = pcy + 0.5f * ph_;
    boxes_f[j * 4 + 0] = b0; boxes_f[j * 4 + 1] = b1;
    boxes_f[j * 4 + 2] = b2; boxes_f[j * 4 + 3] = b3;
    float off = (float)k * 10000.0f;
    boxes_off[j * 4 + 0] = b0 + off; boxes_off[j * 4 + 1] = b1 + off;
    boxes_off[j * 4 + 2] = b2 + off; boxes_off[j * 4 + 3] = b3 + off;
    scores_f[j] = s;
    cand[j] = (s > 0.05f) ? s : -1.0f;
  }
}

// ---------------- 6. Sequential NMS (single workgroup, deterministic argmax) ----------------
__global__ __launch_bounds__(1024)
void nms_kernel(const float* __restrict__ boxes_f,
                const float* __restrict__ boxes_off,
                float* __restrict__ cand,
                float* __restrict__ out) {
  __shared__ float rs[1024];
  __shared__ int   ri[1024];
  __shared__ float keep_s[MAXDET];
  __shared__ int   keep_i[MAXDET];
  __shared__ float bbx[4];
  int t = threadIdx.x;

  for (int it = 0; it < MAXDET; ++it) {
    // argmax over cand, lowest index wins on ties (matches jnp.argmax)
    float bs = cand[t];
    int   bi = t;
    for (int j = t + 1024; j < NCAND; j += 1024) {
      float s = cand[j];
      if (s > bs) { bs = s; bi = j; }
    }
    rs[t] = bs; ri[t] = bi;
    __syncthreads();
    for (int off = 512; off > 0; off >>= 1) {
      if (t < off) {
        float s2 = rs[t + off]; int i2 = ri[t + off];
        if (s2 > rs[t] || (s2 == rs[t] && i2 < ri[t])) { rs[t] = s2; ri[t] = i2; }
      }
      __syncthreads();
    }
    int ib = ri[0];
    if (t == 0) {
      keep_s[it] = rs[0];
      keep_i[it] = ib;
      bbx[0] = boxes_off[ib * 4 + 0]; bbx[1] = boxes_off[ib * 4 + 1];
      bbx[2] = boxes_off[ib * 4 + 2]; bbx[3] = boxes_off[ib * 4 + 3];
    }
    __syncthreads();
    float B0 = bbx[0], B1 = bbx[1], B2 = bbx[2], B3 = bbx[3];
    float areaB = (B2 - B0) * (B3 - B1);
    for (int j = t; j < NCAND; j += 1024) {
      float a0 = boxes_off[j * 4 + 0], a1 = boxes_off[j * 4 + 1];
      float a2 = boxes_off[j * 4 + 2], a3 = boxes_off[j * 4 + 3];
      float ix1 = fmaxf(B0, a0), iy1 = fmaxf(B1, a1);
      float ix2 = fminf(B2, a2), iy2 = fminf(B3, a3);
      float inter = fmaxf(ix2 - ix1, 0.0f) * fmaxf(iy2 - iy1, 0.0f);
      float areaA = (a2 - a0) * (a3 - a1);
      float iou = inter / (areaB + areaA - inter + 1e-9f);
      if (iou > 0.5f) cand[j] = NEG_INF;
    }
    __syncthreads();
  }

  if (t < MAXDET) {
    float sc = keep_s[t];
    int idx = keep_i[t];
    bool valid = sc > 0.05f;
    float mul = valid ? 1.0f : 0.0f;
    out[t * 4 + 0] = boxes_f[idx * 4 + 0] * mul;
    out[t * 4 + 1] = boxes_f[idx * 4 + 1] * mul;
    out[t * 4 + 2] = boxes_f[idx * 4 + 2] * mul;
    out[t * 4 + 3] = boxes_f[idx * 4 + 3] * mul;
    out[4 * MAXDET + t] = valid ? sc : 0.0f;
    ((int*)out)[5 * MAXDET + t] = valid ? (idx % NFG) + 1 : 0;
  }
}

// ---------------- Host-side orchestration ----------------
extern "C" void kernel_launch(void* const* d_in, const int* in_sizes, int n_in,
                              void* d_out, int out_size, void* d_ws, size_t ws_size,
                              hipStream_t stream) {
  (void)in_sizes; (void)n_in; (void)out_size; (void)ws_size;
  const float* fm        = (const float*)d_in[0];
  const float* proposals = (const float*)d_in[1];
  const float* w1        = (const float*)d_in[2];
  const float* b1        = (const float*)d_in[3];
  const float* w2        = (const float*)d_in[4];
  const float* b2        = (const float*)d_in[5];
  const float* wc        = (const float*)d_in[6];
  const float* bc        = (const float*)d_in[7];
  const float* wb        = (const float*)d_in[8];
  const float* bb        = (const float*)d_in[9];

  char* ws = (char*)d_ws;
  size_t off = 0;
  auto alloc = [&](size_t bytes) -> char* {
    char* p = ws + off;
    off += (bytes + 255) & ~(size_t)255;
    return p;
  };
  unsigned short* A1  = (unsigned short*)alloc((size_t)NPAD * FIN * 2);   // pooled, A-packed
  unsigned short* B1  = (unsigned short*)alloc((size_t)FIN * H1 * 2);     // w1, B-packed
  unsigned short* X1  = (unsigned short*)alloc((size_t)NPAD * H1 * 2);    // fc1 out, A-packed
  unsigned short* B2  = (unsigned short*)alloc((size_t)H1 * H1 * 2);      // w2, B-packed
  float* X2   = (float*)alloc((size_t)NPAD * H1 * 4);                     // fc2 out, fp32 rows
  float* CLS  = (float*)alloc((size_t)NPROP * NCLS * 4);
  float* BOX  = (float*)alloc((size_t)NPROP * NCLS * 4 * 4);
  float* SCF  = (float*)alloc((size_t)NCAND * 4);
  float* BF   = (float*)alloc((size_t)NCAND * 4 * 4);
  float* BOFF = (float*)alloc((size_t)NCAND * 4 * 4);
  float* CAND = (float*)alloc((size_t)NCAND * 4);

  // 1. ROI align -> bf16 A-fragments (rows >= 1000 zeroed)
  roi_align_pack_kernel<<<(NPAD * FIN) / 256, 256, 0, stream>>>(fm, proposals, A1);

  // 2. Weight packing
  pack_b_kernel<<<(FIN * H1 + 255) / 256, 256, 0, stream>>>(w1, B1, FIN, H1);
  pack_b_kernel<<<(H1 * H1 + 255) / 256, 256, 0, stream>>>(w2, B2, H1, H1);

  // 3. FC1: [1024,12544] x [12544,1024], bias+ReLU, re-pack as A for FC2
  {
    int Kt = FIN / 32, Nt = H1 / 16;
    int blocks = (NPAD / 128) * (Nt / 4);      // 8 * 16 = 128
    gemm_wmma_kernel<<<blocks, 256, 0, stream>>>(
        A1, B1, b1, Kt, Nt, X1, H1 / 32, nullptr, 0, 1);
  }
  // 4. FC2: [1024,1024] x [1024,1024], bias+ReLU, fp32 rows out
  {
    int Kt = H1 / 32, Nt = H1 / 16;
    int blocks = (NPAD / 128) * (Nt / 4);
    gemm_wmma_kernel<<<blocks, 256, 0, stream>>>(
        X1, B2, b2, Kt, Nt, nullptr, 0, X2, H1, 1);
  }

  // 5. Heads
  head_kernel<<<NPROP, 128, 0, stream>>>(X2, wc, bc, wb, bb, CLS, BOX);

  // 6. Softmax + decode + candidate init
  postproc_kernel<<<(NPROP + 255) / 256, 256, 0, stream>>>(
      CLS, BOX, proposals, SCF, BF, BOFF, CAND);

  // 7. Sequential NMS + final output
  nms_kernel<<<1, 1024, 0, stream>>>(BF, BOFF, CAND, (float*)d_out);
}
